// BidirectionalAttentionV2_40681930228013
// MI455X (gfx1250) — matrix operand order repaired and checked by reading, not compiled, not run on hardware
//
#include <hip/hip_runtime.h>

typedef __attribute__((ext_vector_type(16))) _Float16 v16h;
typedef __attribute__((ext_vector_type(8)))  _Float16 v8h;
typedef __attribute__((ext_vector_type(8)))  float    v8f;

#define TT 1024
#define NN 256
#define QROWS 128
#define KROWS 32
#define NWAVES 8

#define DQ 264   // halves per q/k row in LDS (256 + 8 pad) -> 528B, 16B aligned
#define DV 40    // halves per v^T row (32 keys + 8 pad) -> 80B
#define DP 40    // halves per P row

#define SM_K   0
#define SM_VT  (32 * DQ)                    // 8448
#define SM_P   (32 * DQ + 256 * DV)         // 18688
#define SM_TOT (32 * DQ + 256 * DV + NWAVES * 16 * DP)  // 23808 halves = 47.6 KB
// phase-1 query staging uses [0, 64*DQ) = 16896 halves, also < SM_TOT

__device__ __forceinline__ v16h ld16(const _Float16* p0, const _Float16* p1) {
  v8h a = *(const v8h*)p0;
  v8h b = *(const v8h*)p1;
  return __builtin_shufflevector(a, b, 0,1,2,3,4,5,6,7,8,9,10,11,12,13,14,15);
}

__device__ __forceinline__ v8f wmma16(v16h a, v16h b, v8f c) {
  return __builtin_amdgcn_wmma_f32_16x16x32_f16(false, a, false, b, (short)0, c,
                                                false, false);
}

__launch_bounds__(256, 1)
__global__ void vbdh_flash_attn(const float* __restrict__ Q,
                                const float* __restrict__ V,
                                const float* __restrict__ F,
                                float* __restrict__ O) {
  __shared__ __align__(16) _Float16 smem[SM_TOT];

  const int tid  = threadIdx.x;
  const int lane = tid & 31;
  const int wv   = tid >> 5;
  const int l15  = lane & 15;
  const int h0   = (lane >= 16) ? 8 : 0;    // A-operand half-offset (halves)
  const int hb   = (lane >= 16) ? 8 : 0;    // B-operand half-offset (halves)

  const int qblocks = TT / QROWS;           // 8
  const int bh      = blockIdx.x / qblocks; // 0..95
  const int qbase   = (blockIdx.x % qblocks) * QROWS;
  const float* Qb = Q + (size_t)bh * TT * NN;
  const float* Vb = V + (size_t)bh * TT * NN;

  const float SSCALE = 1.4426950408889634f / 16.0f;  // log2(e) / sqrt(N)
  const float TWOPI  = 6.28318530717958647f;

  // ---------- Phase 1: RoPE queries -> LDS f16 (pre-scaled), load A operand ----------
  v16h qa[8];
  for (int rnd = 0; rnd < 2; ++rnd) {
    #pragma unroll
    for (int i = 0; i < 32; ++i) {
      int pe   = i * 256 + tid;        // pair id over 64 rows x 128 pairs
      int row  = pe >> 7;
      int col  = (pe & 127) * 2;
      int grow = qbase + rnd * 64 + row;
      const float* qp = Qb + (size_t)grow * NN + col;
      float q0 = qp[0], q1 = qp[1];
      float ph = (float)grow * F[col];
      ph -= truncf(ph);                // phases mod 1 (non-negative)
      float sn, cs;
      __sincosf(ph * TWOPI, &sn, &cs);
      smem[row * DQ + col]     = (_Float16)((q0 * cs - q1 * sn) * SSCALE);
      smem[row * DQ + col + 1] = (_Float16)((q1 * cs + q0 * sn) * SSCALE);
    }
    __syncthreads();
    if ((wv >> 2) == rnd) {            // waves 0-3 read round 0, waves 4-7 round 1
      const _Float16* qr = smem + ((wv & 3) * 16 + l15) * DQ;
      #pragma unroll
      for (int kk = 0; kk < 8; ++kk)
        qa[kk] = ld16(qr + kk * 32 + h0, qr + kk * 32 + 16 + h0);
    }
    __syncthreads();
  }

  // ---------- flash-attention state ----------
  const v8f vzero = {};
  v8f acc[16];
  #pragma unroll
  for (int c = 0; c < 16; ++c) acc[c] = vzero;
  float m[8], l[8];
  #pragma unroll
  for (int r = 0; r < 8; ++r) { m[r] = -1.0e30f; l[r] = 0.0f; }

  _Float16* sk  = smem + SM_K;
  _Float16* svt = smem + SM_VT;
  _Float16* sp  = smem + SM_P + wv * (16 * DP);

  #pragma unroll 1
  for (int kt = 0; kt < TT; kt += KROWS) {
    // ---- stage 32 RoPE'd key rows (unscaled f16), row-major ----
    #pragma unroll
    for (int i = 0; i < 16; ++i) {
      int pe   = i * 256 + tid;
      int krow = pe >> 7;
      int col  = (pe & 127) * 2;
      int grow = kt + krow;
      const float* kp = Qb + (size_t)grow * NN + col;
      float q0 = kp[0], q1 = kp[1];
      float ph = (float)grow * F[col];
      ph -= truncf(ph);
      float sn, cs;
      __sincosf(ph * TWOPI, &sn, &cs);
      sk[krow * DQ + col]     = (_Float16)(q0 * cs - q1 * sn);
      sk[krow * DQ + col + 1] = (_Float16)(q1 * cs + q0 * sn);
    }
    // ---- stage V tile transposed: svt[col][key] ----
    #pragma unroll
    for (int i = 0; i < 32; ++i) {
      float v = Vb[(size_t)(kt + i) * NN + tid];
      svt[tid * DV + i] = (_Float16)v;
    }
    if (kt + KROWS < TT) {
      __builtin_prefetch(Qb + (size_t)(kt + KROWS) * NN + tid, 0, 1);
      __builtin_prefetch(Vb + (size_t)(kt + KROWS) * NN + tid, 0, 1);
    }
    __syncthreads();

    // ---- S = (scaled QR) . KR^T : 16 queries x 32 keys ----
    v8f Sa = vzero, Sb = vzero;
    #pragma unroll
    for (int kk = 0; kk < 8; ++kk) {
      const _Float16* kb = sk + l15 * DQ + kk * 32 + hb;
      v16h b0 = ld16(kb, kb + 16);                       // keys kt+0..15
      v16h b1 = ld16(kb + 16 * DQ, kb + 16 * DQ + 16);   // keys kt+16..31
      Sa = wmma16(qa[kk], b0, Sa);
      Sb = wmma16(qa[kk], b1, Sb);
    }

    // ---- online softmax (log2 domain) ----
    float alph[8];
    #pragma unroll
    for (int r = 0; r < 8; ++r) {
      float mx = fmaxf(Sa[r], Sb[r]);
      mx = fmaxf(mx, __shfl_xor(mx, 1));
      mx = fmaxf(mx, __shfl_xor(mx, 2));
      mx = fmaxf(mx, __shfl_xor(mx, 4));
      mx = fmaxf(mx, __shfl_xor(mx, 8));
      float mn = fmaxf(m[r], mx);
      float a  = exp2f(m[r] - mn);
      float pa = exp2f(Sa[r] - mn);
      float pb = exp2f(Sb[r] - mn);
      Sa[r] = pa; Sb[r] = pb;
      float rs = pa + pb;
      rs += __shfl_xor(rs, 1);
      rs += __shfl_xor(rs, 2);
      rs += __shfl_xor(rs, 4);
      rs += __shfl_xor(rs, 8);
      l[r] = l[r] * a + rs;
      m[r] = mn;
      alph[r] = a;
    }
    #pragma unroll
    for (int c = 0; c < 16; ++c) {
      #pragma unroll
      for (int r = 0; r < 8; ++r) acc[c][r] *= alph[r];
    }

    // ---- P (C-layout f32) -> wave-private LDS f16, reload in A layout ----
    #pragma unroll
    for (int r = 0; r < 8; ++r) {
      sp[(r + h0) * DP + l15]      = (_Float16)Sa[r];
      sp[(r + h0) * DP + 16 + l15] = (_Float16)Sb[r];
    }
    asm volatile("s_wait_dscnt 0x0" ::: "memory");  // same-wave LDS RAW, lockstep lanes
    const _Float16* pbp = sp + l15 * DP;
    v16h ap = ld16(pbp + h0, pbp + 16 + h0);

    // ---- O += P . V ----
    #pragma unroll
    for (int c = 0; c < 16; ++c) {
      const _Float16* vb = svt + (c * 16 + l15) * DV + hb;
      v16h bv = ld16(vb, vb + 16);
      acc[c] = wmma16(ap, bv, acc[c]);
    }
    __syncthreads();  // protect sk/svt before next tile's staging
  }

  // ---------- epilogue: O /= l, store ----------
  float* ob = O + ((size_t)bh * TT + qbase + wv * 16) * NN;
  #pragma unroll
  for (int r = 0; r < 8; ++r) {
    float inv = 1.0f / l[r];
    int mr = r + h0;
    #pragma unroll
    for (int c = 0; c < 16; ++c)
      ob[(size_t)mr * NN + c * 16 + l15] = acc[c][r] * inv;
  }
}

extern "C" void kernel_launch(void* const* d_in, const int* in_sizes, int n_in,
                              void* d_out, int out_size, void* d_ws, size_t ws_size,
                              hipStream_t stream) {
  const float* Q = (const float*)d_in[0];
  const float* V = (const float*)d_in[1];
  const float* F = (const float*)d_in[2];
  float* out = (float*)d_out;
  (void)in_sizes; (void)n_in; (void)out_size; (void)d_ws; (void)ws_size;

  const int BH = 8 * 12;
  dim3 grid(BH * (TT / QROWS));   // 96 * 8 = 768 blocks
  dim3 block(256);                // 8 waves (wave32)
  vbdh_flash_attn<<<grid, block, 0, stream>>>(Q, V, F, out);
}
